// CFConv_64888365908355
// MI455X (gfx1250) — matrix operand order repaired and checked by reading, not compiled, not run on hardware
//
#include <hip/hip_runtime.h>

typedef __attribute__((ext_vector_type(16))) _Float16 v16h;
typedef __attribute__((ext_vector_type(8)))  _Float16 v8h;
typedef __attribute__((ext_vector_type(8)))  float    v8f;
typedef __attribute__((ext_vector_type(4)))  unsigned v4u;
typedef __attribute__((ext_vector_type(8)))  int      v8i;
typedef __attribute__((ext_vector_type(4)))  int      v4i;

#define B_  16
#define N_  512
#define K_  64
#define G_  32
#define F_  64

#if __has_builtin(__builtin_amdgcn_tensor_load_to_lds)
#define USE_TDM 1
#else
#define USE_TDM 0
#endif

__device__ __forceinline__ unsigned pack2(float a, float b) {
    union { _Float16 h[2]; unsigned u; } x;
    x.h[0] = (_Float16)a; x.h[1] = (_Float16)b;
    return x.u;
}

__launch_bounds__(128, 1)
__global__ void cfconv_wmma_kernel(const float* __restrict__ features,
                                   const float* __restrict__ rbf,
                                   const long long* __restrict__ nbr_list,
                                   const float* __restrict__ W1,
                                   const float* __restrict__ b1,
                                   const float* __restrict__ W2,
                                   const float* __restrict__ b2,
                                   const float* __restrict__ nbr_filter,
                                   float* __restrict__ out_agg,
                                   float* __restrict__ out_attn) {
    // --- LDS ---
    __shared__ __align__(32) unsigned w1f[4][32][8];        //  4 KB: W1 B-fragments
    __shared__ __align__(32) unsigned w2f[2][4][32][8];     //  8 KB: W2 B-fragments
    __shared__ __align__(32) _Float16 hbuf[4][K_ * F_];     // 32 KB: per-wave rbf(f32)->rbf_norm(f16)->h->filter
    __shared__ float wbuf[4][K_];                           //  1 KB: softmax weights
    __shared__ int   ibuf[4][K_];                           //  1 KB: neighbor indices
    __shared__ float sfilt[F_], sb1[F_], sb2[F_];

    const int tid  = threadIdx.x;
    const int lane = tid & 31;
    const int wv   = tid >> 5;

    const int node = blockIdx.x * 4 + wv;       // one node per wave
    const int b    = node >> 9;                 // node / 512
    _Float16* hb   = hbuf[wv];

    // ---- Issue TDM DMA of this wave's rbf tile (64x32 f32 = 8 KB) into LDS ----
#if USE_TDM
    {
        const unsigned lds_addr = (unsigned)(uintptr_t)hb;              // LDS aperture: low 32 bits = LDS offset
        const unsigned long long ga =
            (unsigned long long)(uintptr_t)(rbf + (size_t)node * (K_ * G_));
        // D# group 0: count=1 (valid), lds_addr, global_addr[56:0], type=2 ("image")
        v4u g0 = { 1u,
                   lds_addr,
                   (unsigned)ga,
                   ((unsigned)((ga >> 32) & 0x01FFFFFFull)) | 0x80000000u };
        // D# group 1: workgroup_mask=0, data_size=2 (4B);
        // tensor_dim0 = 2048 elems, tensor_dim1 = 1, tile_dim0 = 2048 (1D copy),
        // tile_dim1/2 = 0 (unused), tensor_dim0_stride = 2048, tensor_dim1_stride = 0
        v8i g1 = { 0x00020000,            // [17:16] data_size=2
                   0x08000000,            // [63:48] tensor_dim0.lo16 = 0x0800
                   0x00010000,            // [79:64] dim0.hi16=0, [95:80] tensor_dim1.lo16 = 1
                   0x08000000,            // [111:96] dim1.hi16=0, [127:112] tile_dim0 = 0x0800
                   0,                     // tile_dim1 = 0, tile_dim2 = 0
                   2048,                  // tensor_dim0_stride.lo32
                   0, 0 };
        v4i gz4 = { 0, 0, 0, 0 };
        v8i gz8 = { 0, 0, 0, 0, 0, 0, 0, 0 };
        __builtin_amdgcn_tensor_load_to_lds(g0, g1, gz4, gz4, gz8, 0);
    }
#endif

    // ---- Stage weights / biases / nbr_filter (overlaps the TDM transfer) ----
    #pragma unroll
    for (int i = tid; i < 4 * 32 * 8; i += 128) {
        int j = i >> 8;
        int L = (i >> 3) & 31;
        int v = i & 7;
        int col = j * 16 + (L & 15);
        int k0  = ((L >> 4) << 3) + (v << 1) + ((v >> 2) << 3);
        w1f[j][L][v] = pack2(W1[k0 * F_ + col], W1[(k0 + 1) * F_ + col]);
    }
    #pragma unroll
    for (int i = tid; i < 2 * 4 * 32 * 8; i += 128) {
        int s = i >> 10;
        int j = (i >> 8) & 3;
        int L = (i >> 3) & 31;
        int v = i & 7;
        int col = j * 16 + (L & 15);
        int k0  = s * 32 + ((L >> 4) << 3) + (v << 1) + ((v >> 2) << 3);
        w2f[s][j][L][v] = pack2(W2[k0 * F_ + col], W2[(k0 + 1) * F_ + col]);
    }
    if (tid < F_) { sfilt[tid] = nbr_filter[tid]; sb1[tid] = b1[tid]; sb2[tid] = b2[tid]; }

    // ---- Early neighbor-index load + feature-row prefetch (hides gather latency) ----
    const long long* nl = nbr_list + (size_t)node * K_;
    const float*     fb = features + (size_t)b * N_ * F_;
    const int nbr0 = (int)nl[lane];
    const int nbr1 = (int)nl[lane + 32];
    ibuf[wv][lane]      = nbr0;
    ibuf[wv][lane + 32] = nbr1;
    __builtin_prefetch(fb + (size_t)nbr0 * F_, 0, 0);   // -> global_prefetch_b8
    __builtin_prefetch(fb + (size_t)nbr1 * F_, 0, 0);

    __syncthreads();    // weights staged for all waves

    // ---- Phase A: normalize rbf rows, store f16 row-major [64][32] in hb ----
    {
#if USE_TDM
        __builtin_amdgcn_s_wait_tensorcnt(0);           // TDM tile landed in LDS
        const float* rb = (const float*)hb;             // raw f32 tile in LDS
#else
        const float* rb = rbf + (size_t)node * (K_ * G_);
#endif
        #pragma unroll
        for (int rr = 0; rr < 2; ++rr) {
            int row = lane + rr * 32;
            const float4* rp = (const float4*)(rb + row * G_);
            float4 v4[8];
            float ss = 0.f;
            #pragma unroll
            for (int i = 0; i < 8; ++i) {
                v4[i] = rp[i];
                ss += v4[i].x * v4[i].x + v4[i].y * v4[i].y +
                      v4[i].z * v4[i].z + v4[i].w * v4[i].w;
            }
            float inv = 1.0f / (sqrtf(ss) + 1e-8f);
            unsigned* dst = (unsigned*)(hb + row * G_);
            #pragma unroll
            for (int i = 0; i < 8; ++i) {
                dst[i * 2 + 0] = pack2(v4[i].x * inv, v4[i].y * inv);
                dst[i * 2 + 1] = pack2(v4[i].z * inv, v4[i].w * inv);
            }
        }
    }

    const int arow  = lane & 15;            // A-fragment row within 16-row tile
    const int khalf = (lane >> 4) * 8;      // A-fragment K sub-block
    const int crow0 = (lane >> 4) * 8;      // C-fragment row base
    const int ccol  = lane & 15;            // C-fragment column

    // ---- GEMM1: h = tanh(rbf_norm[64,32] @ W1[32,64] + b1) via v_wmma_f32_16x16x32_f16 ----
    v8f acc[4][4];
    #pragma unroll
    for (int j = 0; j < 4; ++j) {
        float bc = sb1[j * 16 + ccol];
        #pragma unroll
        for (int m = 0; m < 4; ++m)
            #pragma unroll
            for (int e = 0; e < 8; ++e) acc[m][j][e] = bc;   // fold bias into C
    }
    #pragma unroll
    for (int m = 0; m < 4; ++m) {
        const _Float16* ap = hb + (m * 16 + arow) * G_ + khalf;
        v8h lo = *(const v8h*)ap;
        v8h hi = *(const v8h*)(ap + 16);
        v16h a = __builtin_shufflevector(lo, hi, 0,1,2,3,4,5,6,7,8,9,10,11,12,13,14,15);
        #pragma unroll
        for (int j = 0; j < 4; ++j) {
            v16h bf = *(const v16h*)(&w1f[j][lane][0]);
            acc[m][j] = __builtin_amdgcn_wmma_f32_16x16x32_f16(
                false, a, false, bf, (short)0, acc[m][j], false, false);
        }
    }
    // tanh + store h (f16, 64x64 row-major) over the rbf staging buffer
    #pragma unroll
    for (int m = 0; m < 4; ++m)
        #pragma unroll
        for (int j = 0; j < 4; ++j)
            #pragma unroll
            for (int r = 0; r < 8; ++r)
                hb[(m * 16 + crow0 + r) * F_ + j * 16 + ccol] = (_Float16)tanhf(acc[m][j][r]);

    // ---- GEMM2: filter = h[64,64] @ W2[64,64] + b2 ----
    v8f acc2[4][4];
    #pragma unroll
    for (int j = 0; j < 4; ++j) {
        float bc = sb2[j * 16 + ccol];
        #pragma unroll
        for (int m = 0; m < 4; ++m)
            #pragma unroll
            for (int e = 0; e < 8; ++e) acc2[m][j][e] = bc;
    }
    #pragma unroll
    for (int s = 0; s < 2; ++s) {
        #pragma unroll
        for (int m = 0; m < 4; ++m) {
            const _Float16* ap = hb + (m * 16 + arow) * F_ + s * 32 + khalf;
            v8h lo = *(const v8h*)ap;
            v8h hi = *(const v8h*)(ap + 16);
            v16h a = __builtin_shufflevector(lo, hi, 0,1,2,3,4,5,6,7,8,9,10,11,12,13,14,15);
            #pragma unroll
            for (int j = 0; j < 4; ++j) {
                v16h bf = *(const v16h*)(&w2f[s][j][lane][0]);
                acc2[m][j] = __builtin_amdgcn_wmma_f32_16x16x32_f16(
                    false, a, false, bf, (short)0, acc2[m][j], false, false);
            }
        }
    }
    // store filter (f16) over h — h fully consumed by GEMM2 reads above
    #pragma unroll
    for (int m = 0; m < 4; ++m)
        #pragma unroll
        for (int j = 0; j < 4; ++j)
            #pragma unroll
            for (int r = 0; r < 8; ++r)
                hb[(m * 16 + crow0 + r) * F_ + j * 16 + ccol] = (_Float16)acc2[m][j][r];

    // ---- Phase C: scores (k-parallel), wave32 softmax ----
    float sc[2];
    #pragma unroll
    for (int rr = 0; rr < 2; ++rr) {
        int k   = lane + rr * 32;
        int nbr = rr ? nbr1 : nbr0;
        const float4*   fr = (const float4*)(fb + (size_t)nbr * F_);
        const _Float16* fl = hb + k * F_;
        float s = 0.f;
        #pragma unroll
        for (int i = 0; i < 16; ++i) {
            float4 fv = fr[i];
            s += fv.x * (float)fl[i * 4 + 0] * sfilt[i * 4 + 0];
            s += fv.y * (float)fl[i * 4 + 1] * sfilt[i * 4 + 1];
            s += fv.z * (float)fl[i * 4 + 2] * sfilt[i * 4 + 2];
            s += fv.w * (float)fl[i * 4 + 3] * sfilt[i * 4 + 3];
        }
        sc[rr] = s;
    }
    float mx = fmaxf(sc[0], sc[1]);
    #pragma unroll
    for (int off = 16; off > 0; off >>= 1) mx = fmaxf(mx, __shfl_xor(mx, off, 32));
    float e0 = __expf(sc[0] - mx), e1 = __expf(sc[1] - mx);
    float sm = e0 + e1;
    #pragma unroll
    for (int off = 16; off > 0; off >>= 1) sm += __shfl_xor(sm, off, 32);
    float rinv = 1.0f / sm;
    float aw0 = e0 * rinv, aw1 = e1 * rinv;
    out_attn[(size_t)node * K_ + lane]      = aw0;
    out_attn[(size_t)node * K_ + lane + 32] = aw1;
    wbuf[wv][lane]      = aw0;
    wbuf[wv][lane + 32] = aw1;

    // ---- Phase D: aggregation (f-parallel; coalesced, L2-hot feature reloads) ----
    int f0 = lane * 2;
    float a0 = 0.f, a1 = 0.f;
    for (int k = 0; k < K_; ++k) {
        float wk = wbuf[wv][k];
        int  nbr = ibuf[wv][k];
        float2 fv = *(const float2*)(fb + (size_t)nbr * F_ + f0);
        a0 += wk * fv.x * (float)hb[k * F_ + f0];
        a1 += wk * fv.y * (float)hb[k * F_ + f0 + 1];
    }
    float2 ov; ov.x = a0; ov.y = a1;
    *(float2*)(out_agg + (size_t)node * F_ + f0) = ov;
}

extern "C" void kernel_launch(void* const* d_in, const int* in_sizes, int n_in,
                              void* d_out, int out_size, void* d_ws, size_t ws_size,
                              hipStream_t stream) {
    const float*     features = (const float*)d_in[0];
    const float*     rbf      = (const float*)d_in[1];
    const long long* nl       = (const long long*)d_in[2];
    const float*     W1       = (const float*)d_in[3];
    const float*     b1       = (const float*)d_in[4];
    const float*     W2       = (const float*)d_in[5];
    const float*     b2       = (const float*)d_in[6];
    const float*     nf       = (const float*)d_in[7];

    float* out_agg  = (float*)d_out;
    float* out_attn = out_agg + (size_t)B_ * N_ * F_;

    const int blocks = (B_ * N_) / 4;   // 4 waves (nodes) per 128-thread block
    cfconv_wmma_kernel<<<blocks, 128, 0, stream>>>(
        features, rbf, nl, W1, b1, W2, b2, nf, out_agg, out_attn);
}
